// FocusedMQAttention_9259949490453
// MI455X (gfx1250) — matrix operand order repaired and checked by reading, not compiled, not run on hardware
//
#include <hip/hip_runtime.h>
#include <hip/hip_bf16.h>
#include <stdint.h>

// ---------------------------------------------------------------------------
// FocusedMQAttention for MI455X (gfx1250), wave32 + WMMA bf16 +
// global_load_async_to_lds tile staging.
// ---------------------------------------------------------------------------

typedef __bf16 bf16;
typedef __attribute__((ext_vector_type(16))) __bf16 v16bf;
typedef __attribute__((ext_vector_type(8)))  __bf16 v8bf;
typedef __attribute__((ext_vector_type(8)))  float  v8f;

#define B_   16
#define N_   4096
#define C_   1152
#define H_   16
#define HD_  72
#define S_   64
#define KV2_ 144   // 2*HD

// ---------------------------- WMMA GEMM ------------------------------------
// D[M,N] = A[M,K](bf16) * Bt[N,K](bf16, pre-transposed) + bias[N], D fp32.
// Block: 256 threads = 8 waves (2 along M x 4 along N).
// Tile: BM=128, BN=128, BK=32 (one wmma k-step). Double-buffered LDS filled
// with global_load_async_to_lds_b128 (ASYNCcnt).

#define BM 128
#define BN 128
#define BK 32
#define LDSK 40   // padded row stride (elements); 80B keeps 16B alignment

__global__ __launch_bounds__(256)
void wmma_gemm_bf16(const bf16* __restrict__ A, const bf16* __restrict__ Bt,
                    const float* __restrict__ bias, float* __restrict__ D,
                    int M, int N, int K)
{
    __shared__ bf16 Alds[2][BM][LDSK];
    __shared__ bf16 Blds[2][BN][LDSK];   // [n][k]

    const int tid  = threadIdx.x;
    const int lane = tid & 31;
    const int wave = tid >> 5;
    const int lh   = lane & 15;   // 0..15
    const int sel  = lane >> 4;   // 0 or 1
    const int waveM = (wave & 1) * 64;    // 2 waves along M -> 64 rows each
    const int waveN = (wave >> 1) * 32;   // 4 waves along N -> 32 cols each

    const int row0 = blockIdx.y * BM;
    const int col0 = blockIdx.x * BN;

    // per-thread tile chunk mapping (shared by A and B): 512 chunks of 8 bf16
    const int rA = tid >> 2;          // row 0..63 (+64 on second pass)
    const int kc = (tid & 3) * 8;     // k offset within tile: 0/8/16/24

    v8f acc[4][2];
    #pragma unroll
    for (int i = 0; i < 4; ++i)
        #pragma unroll
        for (int j = 0; j < 2; ++j)
            #pragma unroll
            for (int e = 0; e < 8; ++e) acc[i][j][e] = 0.0f;

    // -- async DMA of one (A,B) tile pair into LDS buffer `buf` --------------
    auto issueTile = [&](int buf, int kt) {
        size_t kbase = (size_t)kt * BK + kc;
        #pragma unroll
        for (int i = 0; i < 2; ++i) {
            int r = rA + i * 64;
            // A chunk: row (row0+r), 8 contiguous bf16 at kbase
            size_t grow = (size_t)row0 + r;
            if (grow >= (size_t)M) grow = 0;          // clamp (result unused)
            const bf16* gA = A + grow * (size_t)K + kbase;
            unsigned la = (unsigned)(uintptr_t)(&Alds[buf][r][kc]);
            asm volatile("global_load_async_to_lds_b128 %0, %1, off"
                         :: "v"(la), "v"(gA) : "memory");
            // B chunk: column (col0+r) of B == row of Bt, 8 contiguous bf16
            int gcol = col0 + r;
            if (gcol >= N) gcol = 0;                  // clamp (cols guarded
                                                      //  at the epilogue)
            const bf16* gB = Bt + (size_t)gcol * (size_t)K + kbase;
            unsigned lb = (unsigned)(uintptr_t)(&Blds[buf][r][kc]);
            asm volatile("global_load_async_to_lds_b128 %0, %1, off"
                         :: "v"(lb), "v"(gB) : "memory");
        }
    };
    auto waitTiles = [&]() {
        asm volatile("s_wait_asynccnt 0x0" ::: "memory");
        __syncthreads();
    };

    // -- fragment loads + wmma ----------------------------------------------
    auto compute = [&](int buf) {
        v16bf afr[4], bfr[2];
        // A 16x32 bf16 layout: lane<16 -> M=lh, K = {0..7, 16..23};
        //                      lane>=16 -> M=lh, K = {8..15, 24..31}
        #pragma unroll
        for (int mt = 0; mt < 4; ++mt) {
            const bf16* base = &Alds[buf][waveM + mt * 16 + lh][sel * 8];
            union { v16bf v; v8bf h[2]; } u;
            u.h[0] = *(const v8bf*)(base);
            u.h[1] = *(const v8bf*)(base + 16);
            afr[mt] = u.v;
        }
        // B 32x16 bf16 layout: lane<16 -> N=lh, K = {0..7, 16..23}; lane>=16 -> K = {8..15, 24..31}
        #pragma unroll
        for (int nt = 0; nt < 2; ++nt) {
            const bf16* base = &Blds[buf][waveN + nt * 16 + lh][sel * 8];
            union { v16bf v; v8bf h[2]; } u;
            u.h[0] = *(const v8bf*)(base);
            u.h[1] = *(const v8bf*)(base + 16);
            bfr[nt] = u.v;
        }
        #pragma unroll
        for (int mt = 0; mt < 4; ++mt)
            #pragma unroll
            for (int nt = 0; nt < 2; ++nt)
                acc[mt][nt] = __builtin_amdgcn_wmma_f32_16x16x32_bf16(
                    false, afr[mt], false, bfr[nt], (short)0, acc[mt][nt],
                    false, false);
    };

    // -- main loop: double buffered, async prefetch of next tile -------------
    const int KT = K / BK;
    issueTile(0, 0);
    waitTiles();
    for (int kt = 0; kt < KT; ++kt) {
        int buf = kt & 1;
        if (kt + 1 < KT) issueTile(buf ^ 1, kt + 1);
        compute(buf);
        if (kt + 1 < KT) waitTiles();
    }

    // -- epilogue: C/D layout: VGPR r -> M = r + sel*8, N = lh ---------------
    #pragma unroll
    for (int mt = 0; mt < 4; ++mt) {
        #pragma unroll
        for (int nt = 0; nt < 2; ++nt) {
            int gcol = col0 + waveN + nt * 16 + lh;
            if (gcol < N) {
                float bb = bias ? bias[gcol] : 0.0f;
                #pragma unroll
                for (int r = 0; r < 8; ++r) {
                    int grow = row0 + waveM + mt * 16 + sel * 8 + r;
                    if (grow < M)
                        D[(size_t)grow * N + gcol] = acc[mt][nt][r] + bb;
                }
            }
        }
    }
}

// ------------------------- helper kernels ----------------------------------

__global__ void zero_f32(float* p, int n)
{
    int i = blockIdx.x * blockDim.x + threadIdx.x;
    if (i < n) p[i] = 0.0f;
}

__global__ void f32_to_bf16(const float* __restrict__ s, bf16* __restrict__ d,
                            size_t n)
{
    size_t stride = (size_t)gridDim.x * blockDim.x;
    for (size_t i = (size_t)blockIdx.x * blockDim.x + threadIdx.x; i < n;
         i += stride)
        d[i] = (bf16)s[i];
}

// s[rows][cols] fp32 -> d[cols][rows] bf16 (transpose during conversion)
__global__ void f32_to_bf16_t(const float* __restrict__ s,
                              bf16* __restrict__ d, int rows, int cols)
{
    size_t n = (size_t)rows * cols;
    size_t stride = (size_t)gridDim.x * blockDim.x;
    for (size_t i = (size_t)blockIdx.x * blockDim.x + threadIdx.x; i < n;
         i += stride) {
        int k = (int)(i % rows);   // inner dim of output = row of source
        int c = (int)(i / rows);   // outer dim of output = col of source
        d[i] = (bf16)s[(size_t)k * cols + c];
    }
}

// group sums of relu(x)^2 and relu(x)^6 over [rows x cols] with row stride ld.
__global__ __launch_bounds__(256)
void focused_sums(const float* __restrict__ src, int innerGroups,
                  size_t outerStride, int innerStride, int ld,
                  int rows, int cols, int blocksPerGroup,
                  float* __restrict__ s2, float* __restrict__ s6)
{
    int g   = blockIdx.x / blocksPerGroup;
    int sub = blockIdx.x % blocksPerGroup;
    int b   = g / innerGroups;
    int h   = g % innerGroups;
    const float* base = src + (size_t)b * outerStride + (size_t)h * innerStride;
    long total = (long)rows * cols;
    float a2 = 0.0f, a6 = 0.0f;
    for (long idx = (long)sub * blockDim.x + threadIdx.x; idx < total;
         idx += (long)blocksPerGroup * blockDim.x) {
        int n = (int)(idx / cols), dcol = (int)(idx % cols);
        float v = base[(size_t)n * ld + dcol];
        float r = v > 0.0f ? v : 0.0f;
        float r2 = r * r, r3 = r2 * r;
        a2 += r2;
        a6 += r3 * r3;
    }
    __shared__ float red2[256], red6[256];
    red2[threadIdx.x] = a2;
    red6[threadIdx.x] = a6;
    __syncthreads();
    for (int s = 128; s > 0; s >>= 1) {
        if ((int)threadIdx.x < s) {
            red2[threadIdx.x] += red2[threadIdx.x + s];
            red6[threadIdx.x] += red6[threadIdx.x + s];
        }
        __syncthreads();
    }
    if (threadIdx.x == 0) {
        atomicAdd(&s2[g], red2[0]);
        atomicAdd(&s6[g], red6[0]);
    }
}

// q <- scale(b,h) * relu(q)^3, in place.
__global__ void focused_apply_q(float* __restrict__ q,
                                const float* __restrict__ s2,
                                const float* __restrict__ s6)
{
    size_t n = (size_t)B_ * N_ * C_;
    size_t stride = (size_t)gridDim.x * blockDim.x;
    for (size_t i = (size_t)blockIdx.x * blockDim.x + threadIdx.x; i < n;
         i += stride) {
        size_t row = i / C_;
        int c = (int)(i % C_);
        int b = (int)(row / N_);
        int h = c / HD_;
        int g = b * H_ + h;
        float sc = sqrtf(s2[g] / fmaxf(s6[g], 1e-30f));
        float v = q[i];
        float r = v > 0.0f ? v : 0.0f;
        q[i] = sc * r * r * r;
    }
}

// kv[b][d][e] = sum_n kf[b][n][d] * v[b][n][e], kf = kscale*relu(k)^3.
__global__ __launch_bounds__(256)
void kv_accum(const float* __restrict__ kvlin, const float* __restrict__ ks2,
              const float* __restrict__ ks6, float* __restrict__ kv)
{
    int b = blockIdx.y;
    int chunk = blockIdx.x;  // 16 chunks of 256 n each
    float sc = sqrtf(ks2[b] / fmaxf(ks6[b], 1e-30f));
    __shared__ float kf[64][HD_];
    __shared__ float vv[64][HD_];
    const int PAIRS = (HD_ * HD_ + 255) / 256;  // 21
    float acc[PAIRS];
    #pragma unroll
    for (int p = 0; p < PAIRS; ++p) acc[p] = 0.0f;

    for (int sub = 0; sub < 4; ++sub) {
        int nbase = chunk * 256 + sub * 64;
        for (int i = threadIdx.x; i < 64 * HD_; i += 256) {
            int n = i / HD_, d = i % HD_;
            const float* row = kvlin + ((size_t)b * N_ + nbase + n) * KV2_;
            float kvp = row[d];
            float r = kvp > 0.0f ? kvp : 0.0f;
            kf[n][d] = sc * r * r * r;
            vv[n][d] = row[HD_ + d];
        }
        __syncthreads();
        for (int p = 0; p < PAIRS; ++p) {
            int idx = threadIdx.x + p * 256;
            if (idx < HD_ * HD_) {
                int d = idx / HD_, e = idx % HD_;
                float a = acc[p];
                #pragma unroll 8
                for (int n = 0; n < 64; ++n) a += kf[n][d] * vv[n][e];
                acc[p] = a;
            }
        }
        __syncthreads();
    }
    for (int p = 0; p < PAIRS; ++p) {
        int idx = threadIdx.x + p * 256;
        if (idx < HD_ * HD_)
            atomicAdd(&kv[(size_t)b * HD_ * HD_ + idx], acc[p]);
    }
}

// depthwise 3x3 SAME conv on v viewed as [B][HD][64][64]; out [B][N][HD].
__global__ void dwconv(const float* __restrict__ kvlin,
                       const float* __restrict__ w,
                       const float* __restrict__ bias,
                       float* __restrict__ vdwc)
{
    size_t total = (size_t)B_ * N_ * HD_;
    size_t stride = (size_t)gridDim.x * blockDim.x;
    for (size_t i = (size_t)blockIdx.x * blockDim.x + threadIdx.x; i < total;
         i += stride) {
        int ch = (int)(i % HD_);
        size_t bn = i / HD_;
        int b = (int)(bn / N_);
        int n = (int)(bn % N_);
        int y = n >> 6, x = n & 63;
        float a = bias[ch];
        #pragma unroll
        for (int ky = 0; ky < 3; ++ky) {
            int yy = y + ky - 1;
            if (yy < 0 || yy >= S_) continue;
            #pragma unroll
            for (int kx = 0; kx < 3; ++kx) {
                int xx = x + kx - 1;
                if (xx < 0 || xx >= S_) continue;
                int nn = yy * S_ + xx;
                a += w[ch * 9 + ky * 3 + kx] *
                     kvlin[((size_t)b * N_ + nn) * KV2_ + HD_ + ch];
            }
        }
        vdwc[i] = a;
    }
}

// y[b][n][h*HD+e] = sum_d q[b][n][h*HD+d]*kv[b][d][e] + vdwc[b][n][e]  (bf16)
__global__ __launch_bounds__(256)
void attn_add(const float* __restrict__ q, const float* __restrict__ kv,
              const float* __restrict__ vdwc, bf16* __restrict__ y)
{
    int b = blockIdx.y;
    int nbase = blockIdx.x * 64;
    __shared__ float kvl[HD_ * HD_];
    for (int i = threadIdx.x; i < HD_ * HD_; i += 256)
        kvl[i] = kv[(size_t)b * HD_ * HD_ + i];
    __syncthreads();

    // 64 n * 16 h = 1024 (n,h) pairs -> 4 per thread
    for (int p = 0; p < 4; ++p) {
        int pair = threadIdx.x + p * 256;
        int n = nbase + (pair >> 4);
        int h = pair & 15;
        const float* qrow = q + ((size_t)b * N_ + n) * C_ + h * HD_;
        float qr[HD_];
        #pragma unroll
        for (int d = 0; d < HD_; ++d) qr[d] = qrow[d];
        const float* vd = vdwc + ((size_t)b * N_ + n) * HD_;
        bf16* yrow = y + ((size_t)b * N_ + n) * C_ + h * HD_;
        for (int e = 0; e < HD_; ++e) {
            float a = 0.0f;
            #pragma unroll 8
            for (int d = 0; d < HD_; ++d) a += qr[d] * kvl[d * HD_ + e];
            yrow[e] = (bf16)(a + vd[e]);
        }
    }
}

// ------------------------------ launcher ------------------------------------

extern "C" void kernel_launch(void* const* d_in, const int* in_sizes, int n_in,
                              void* d_out, int out_size, void* d_ws,
                              size_t ws_size, hipStream_t stream)
{
    const float* x      = (const float*)d_in[0];
    const float* wq_w   = (const float*)d_in[1];
    const float* wq_b   = (const float*)d_in[2];
    const float* wkv_w  = (const float*)d_in[3];
    const float* wkv_b  = (const float*)d_in[4];
    const float* dwc_w  = (const float*)d_in[5];
    const float* dwc_b  = (const float*)d_in[6];
    const float* proj_w = (const float*)d_in[7];
    const float* proj_b = (const float*)d_in[8];
    float* out = (float*)d_out;

    char* ws = (char*)d_ws;
    size_t off = 0;
    auto alloc = [&](size_t bytes) -> void* {
        void* p = ws + off;
        off += (bytes + 255) & ~(size_t)255;
        return p;
    };

    const size_t MN = (size_t)B_ * N_;  // 65536 rows

    bf16*  x_bf    = (bf16*)alloc(MN * C_ * sizeof(bf16));   // reused later as y
    bf16*  wq_bf   = (bf16*)alloc((size_t)C_ * C_ * sizeof(bf16));    // [N][K]
    bf16*  wkv_bf  = (bf16*)alloc((size_t)C_ * KV2_ * sizeof(bf16));  // [N][K]
    bf16*  proj_bf = (bf16*)alloc((size_t)C_ * C_ * sizeof(bf16));    // [N][K]
    float* qf      = (float*)alloc(MN * C_ * sizeof(float));
    float* kvlin   = (float*)alloc(MN * KV2_ * sizeof(float));
    float* vdwc    = (float*)alloc(MN * HD_ * sizeof(float));
    // zero-initialized block (kv matrix + norm accumulators), contiguous:
    size_t zoff = off;
    float* kvm = (float*)alloc((size_t)B_ * HD_ * HD_ * sizeof(float));
    float* qs2 = (float*)alloc(256 * sizeof(float));
    float* qs6 = (float*)alloc(256 * sizeof(float));
    float* ks2 = (float*)alloc(16 * sizeof(float));
    float* ks6 = (float*)alloc(16 * sizeof(float));
    int zcount = (int)((off - zoff) / sizeof(float));

    // 0) zero accumulators
    zero_f32<<<(zcount + 255) / 256, 256, 0, stream>>>((float*)(ws + zoff),
                                                       zcount);

    // 1) fp32 -> bf16 conversions (weights transposed to [N][K])
    f32_to_bf16<<<4096, 256, 0, stream>>>(x, x_bf, MN * C_);
    f32_to_bf16_t<<<1024, 256, 0, stream>>>(wq_w, wq_bf, C_, C_);
    f32_to_bf16_t<<<256, 256, 0, stream>>>(wkv_w, wkv_bf, C_, KV2_);
    f32_to_bf16_t<<<1024, 256, 0, stream>>>(proj_w, proj_bf, C_, C_);

    // 2) q = x @ wq_w + wq_b         [65536 x 1152]
    {
        dim3 grid((C_ + BN - 1) / BN, (int)(MN / BM));
        wmma_gemm_bf16<<<grid, 256, 0, stream>>>(x_bf, wq_bf, wq_b, qf,
                                                 (int)MN, C_, C_);
    }
    // 3) kv_lin = x @ wkv_w + wkv_b  [65536 x 144]
    {
        dim3 grid((KV2_ + BN - 1) / BN, (int)(MN / BM));
        wmma_gemm_bf16<<<grid, 256, 0, stream>>>(x_bf, wkv_bf, wkv_b, kvlin,
                                                 (int)MN, KV2_, C_);
    }

    // 4) focused norms for q: 256 groups (b,h), 2 blocks each
    focused_sums<<<256 * 2, 256, 0, stream>>>(qf, H_, (size_t)N_ * C_, HD_,
                                              C_, N_, HD_, 2, qs2, qs6);
    // 5) focused norms for k: 16 groups (b), 16 blocks each
    focused_sums<<<16 * 16, 256, 0, stream>>>(kvlin, 1, (size_t)N_ * KV2_, 0,
                                              KV2_, N_, HD_, 16, ks2, ks6);

    // 6) q <- focused(q) in place
    focused_apply_q<<<2048, 256, 0, stream>>>(qf, qs2, qs6);

    // 7) kv = focused(k)^T @ v per batch
    {
        dim3 grid(16, B_);
        kv_accum<<<grid, 256, 0, stream>>>(kvlin, ks2, ks6, kvm);
    }

    // 8) depthwise conv residual
    dwconv<<<2048, 256, 0, stream>>>(kvlin, dwc_w, dwc_b, vdwc);

    // 9) y = q @ kv + v_dwc, bf16, reusing x_bf storage (x dead after GEMMs)
    {
        dim3 grid(N_ / 64, B_);
        attn_add<<<grid, 256, 0, stream>>>(qf, kvm, vdwc, x_bf);
    }

    // 10) out = y @ proj_w + proj_b
    {
        dim3 grid((C_ + BN - 1) / BN, (int)(MN / BM));
        wmma_gemm_bf16<<<grid, 256, 0, stream>>>(x_bf, proj_bf, proj_b, out,
                                                 (int)MN, C_, C_);
    }
    (void)in_sizes; (void)n_in; (void)out_size; (void)ws_size;
}
